// TriangleMultiplicationOpti_55310588838175
// MI455X (gfx1250) — matrix operand order repaired and checked by reading, not compiled, not run on hardware
//
#include <hip/hip_runtime.h>
#include <hip/hip_bf16.h>
#include <stdint.h>

// TriangleMultiplicationOutgoing for MI455X (gfx1250, wave32, WMMA bf16).
// Pipeline:
//   K0: convert 6 weight matrices fp32 -> bf16 into ws
//   K1: fused LN + 5 projections (WMMA) -> left_t/right_t (channel-major bf16), gate (row-major bf16)
//   K2: 128 x per-channel 768^3 NT-GEMM (WMMA bf16, fp32 acc), double-buffered
//       async global->LDS staging -> tri fp32 channel-major
//   K3: LN over C + op projection (WMMA) * gate -> out fp32
// ws requirement: 196608 + 3*150994944 + 301989888 = ~755.2 MB

#define N_RES 768
#define CDIM  128
#define NN    (N_RES * N_RES)   // 589824
#define LN_EPS 1e-5f

typedef __attribute__((ext_vector_type(16))) __bf16 v16bf;
typedef __attribute__((ext_vector_type(8)))  __bf16 v8bf;
typedef __attribute__((ext_vector_type(8)))  float  v8f;
typedef __attribute__((ext_vector_type(4)))  int    v4i;

// ---- CDNA5 async global->LDS path (builtin confirmed present; param0 is
// ---- global (AS1) v4i*, param1 is LDS (AS3) v4i*, then imm offset, imm cpol) ----
#if defined(__AMDGCN__) && __has_builtin(__builtin_amdgcn_global_load_async_to_lds_b128) && \
    __has_builtin(__builtin_amdgcn_s_wait_asynccnt)
#define HAVE_ASYNC 1
#else
#define HAVE_ASYNC 0
#endif

__device__ __forceinline__ void async_cp16(void* lds_dst, const void* gsrc) {
#if HAVE_ASYNC
    typedef __attribute__((address_space(1))) v4i GV4;
    typedef __attribute__((address_space(3))) v4i LV4;
    __builtin_amdgcn_global_load_async_to_lds_b128((GV4*)gsrc, (LV4*)lds_dst, 0, 0);
#else
    *reinterpret_cast<uint4*>(lds_dst) = *reinterpret_cast<const uint4*>(gsrc);
#endif
}
__device__ __forceinline__ void async_wait0() {
#if HAVE_ASYNC
    __builtin_amdgcn_s_wait_asynccnt(0);
#endif
}

__device__ __forceinline__ unsigned short f2bf(float f) {
    union { float f; unsigned int u; } v; v.f = f;
    unsigned int r = v.u + 0x7FFFu + ((v.u >> 16) & 1u);   // round-to-nearest-even
    return (unsigned short)(r >> 16);
}
__device__ __forceinline__ float bf2f(unsigned short h) {
    union { unsigned int u; float f; } v; v.u = ((unsigned int)h) << 16;
    return v.f;
}

// Build a v16bf WMMA fragment from two 16-byte chunks (LDS or global).
__device__ __forceinline__ v16bf make_frag(const unsigned short* p0, const unsigned short* p1) {
    v8bf lo = *reinterpret_cast<const v8bf*>(p0);
    v8bf hi = *reinterpret_cast<const v8bf*>(p1);
    v16bf r;
#pragma unroll
    for (int i = 0; i < 8; ++i) { r[i] = lo[i]; r[i + 8] = hi[i]; }
    return r;
}

#define WMMA_BF16(A, B, ACC) \
    __builtin_amdgcn_wmma_f32_16x16x32_bf16(false, (A), false, (B), (short)0, (ACC), false, false)

// ---------------------------------------------------------------------------
// K0: weight fp32 -> bf16.  Order in Wb: [0]=lp [1]=lg [2]=rp [3]=rg [4]=gl [5]=op
// ---------------------------------------------------------------------------
__global__ void __launch_bounds__(256)
k_convert_weights(const float* __restrict__ lp, const float* __restrict__ lg,
                  const float* __restrict__ rp, const float* __restrict__ rg,
                  const float* __restrict__ gl, const float* __restrict__ op,
                  unsigned short* __restrict__ Wb) {
    int t = blockIdx.x * 256 + threadIdx.x;          // 0 .. 6*16384-1
    const float* srcs[6] = { lp, lg, rp, rg, gl, op };
    int mat = t >> 14;
    int idx = t & 16383;
    Wb[t] = f2bf(srcs[mat][idx]);
}

// ---------------------------------------------------------------------------
// K1: LN + 5 projections.  Block = 256 thr (8 waves), 64 rows of act.
// ---------------------------------------------------------------------------
__global__ void __launch_bounds__(256)
k_ln_proj(const float* __restrict__ act,  const float* __restrict__ mask,
          const float* __restrict__ lnw,  const float* __restrict__ lnb,
          const float* __restrict__ lpb,  const float* __restrict__ lgb,
          const float* __restrict__ rpb,  const float* __restrict__ rgb,
          const float* __restrict__ glb,
          const unsigned short* __restrict__ Wb,
          unsigned short* __restrict__ leftT,
          unsigned short* __restrict__ rightT,
          unsigned short* __restrict__ gate) {
    __shared__ __align__(16) unsigned short sA[64 * 144];     // 18,432 B
    __shared__ __align__(16) unsigned short smemU[128 * 144]; // 36,864 B
    __shared__ float red[2][64][4];                            //  2,048 B

    const int tid  = threadIdx.x;
    const int lane = tid & 31;
    const int wid  = tid >> 5;
    const int r0   = blockIdx.x * 64;

    // ---- LayerNorm of 64 rows (4 threads per row, 32 floats each) ----
    {
        int m = tid >> 2, q = tid & 3;
        const float4* src = reinterpret_cast<const float4*>(act + (size_t)(r0 + m) * CDIM + q * 32);
        float4 vals[8];
        float s = 0.f, ss = 0.f;
#pragma unroll
        for (int j = 0; j < 8; ++j) {
            vals[j] = src[j];
            s  += vals[j].x + vals[j].y + vals[j].z + vals[j].w;
            ss += vals[j].x * vals[j].x + vals[j].y * vals[j].y
                + vals[j].z * vals[j].z + vals[j].w * vals[j].w;
        }
        red[0][m][q] = s; red[1][m][q] = ss;
        __syncthreads();
        float st  = red[0][m][0] + red[0][m][1] + red[0][m][2] + red[0][m][3];
        float sst = red[1][m][0] + red[1][m][1] + red[1][m][2] + red[1][m][3];
        float mu   = st * (1.0f / CDIM);
        float var  = sst * (1.0f / CDIM) - mu * mu;
        float rstd = rsqrtf(var + LN_EPS);
#pragma unroll
        for (int j = 0; j < 8; ++j) {
            int c = q * 32 + j * 4;
            float4 w4 = *reinterpret_cast<const float4*>(lnw + c);
            float4 b4 = *reinterpret_cast<const float4*>(lnb + c);
            sA[m * 144 + c + 0] = f2bf((vals[j].x - mu) * rstd * w4.x + b4.x);
            sA[m * 144 + c + 1] = f2bf((vals[j].y - mu) * rstd * w4.y + b4.y);
            sA[m * 144 + c + 2] = f2bf((vals[j].z - mu) * rstd * w4.z + b4.z);
            sA[m * 144 + c + 3] = f2bf((vals[j].w - mu) * rstd * w4.w + b4.w);
        }
    }

    // Stage one 128x128 bf16 weight matrix into smemU (pitch 144 halves).
    auto stageW = [&](int widx) {
        __syncthreads();   // prior users of smemU / sA writers are done
        int d = tid >> 1, hh = tid & 1;
        const unsigned short* s = Wb + widx * 16384 + d * 128 + hh * 64;
        unsigned short* dst = &smemU[d * 144 + hh * 64];
#pragma unroll
        for (int j = 0; j < 8; ++j) async_cp16(dst + j * 8, s + j * 8);
        async_wait0();
        __syncthreads();
    };

    // Wave wid: rows (wid&3)*16..+15, output cols (wid>>2)*64..+63 (4 n-tiles).
    auto gemm = [&](v8f acc[4]) {
        const int mrow  = (wid & 3) * 16 + (lane & 15);
        const int nbase = (wid >> 2) * 64;
#pragma unroll
        for (int kc = 0; kc < 4; ++kc) {
            int ka = kc * 32 + ((lane < 16) ? 0 : 8);
            v16bf A = make_frag(&sA[mrow * 144 + ka], &sA[mrow * 144 + ka + 16]);
#pragma unroll
            for (int nn = 0; nn < 4; ++nn) {
                int d  = nbase + nn * 16 + (lane & 15);
                int kb = kc * 32 + ((lane < 16) ? 0 : 8);
                v16bf B = make_frag(&smemU[d * 144 + kb], &smemU[d * 144 + kb + 16]);
                acc[nn] = WMMA_BF16(A, B, acc[nn]);
            }
        }
    };

    // Combine proj + sigmoid(gate) * mask, transpose through LDS, write channel-major.
    auto writePair = [&](v8f accP[4], v8f accG[4],
                         const float* __restrict__ pb, const float* __restrict__ gb,
                         unsigned short* __restrict__ dstT) {
        float mvals[8];
#pragma unroll
        for (int e = 0; e < 8; ++e) {
            int mloc = (wid & 3) * 16 + ((lane < 16) ? e : e + 8);
            mvals[e] = mask[r0 + mloc];
        }
        __syncthreads();                       // all waves done reading smemU as weights
        unsigned short* outT = smemU;          // reuse as [128][72] bf16
#pragma unroll
        for (int nn = 0; nn < 4; ++nn) {
            int d  = (wid >> 2) * 64 + nn * 16 + (lane & 15);
            float bp = pb[d], bg = gb[d];
#pragma unroll
            for (int e = 0; e < 8; ++e) {
                int mloc = (wid & 3) * 16 + ((lane < 16) ? e : e + 8);
                float p = accP[nn][e] + bp;
                float g = accG[nn][e] + bg;
                float r = mvals[e] * p * (1.0f / (1.0f + __expf(-g)));
                outT[d * 72 + mloc] = f2bf(r);
            }
        }
        __syncthreads();
        int d = tid >> 1, hh = tid & 1;        // channel-major coalesced store
        const uint4* s = reinterpret_cast<const uint4*>(outT + d * 72 + hh * 32);
        uint4* g4 = reinterpret_cast<uint4*>(dstT + (size_t)d * NN + r0 + hh * 32);
#pragma unroll
        for (int j = 0; j < 4; ++j) g4[j] = s[j];
    };

    v8f z = {};
    v8f accP[4] = { z, z, z, z };
    v8f accG[4] = { z, z, z, z };

    stageW(0); gemm(accP);                 // lp
    stageW(1); gemm(accG);                 // lg
    writePair(accP, accG, lpb, lgb, leftT);

#pragma unroll
    for (int i = 0; i < 4; ++i) { accP[i] = z; accG[i] = z; }
    stageW(2); gemm(accP);                 // rp
    stageW(3); gemm(accG);                 // rg
    writePair(accP, accG, rpb, rgb, rightT);

#pragma unroll
    for (int i = 0; i < 4; ++i) accG[i] = z;
    stageW(4); gemm(accG);                 // gl -> gate, row-major [r][c]
#pragma unroll
    for (int nn = 0; nn < 4; ++nn) {
        int d  = (wid >> 2) * 64 + nn * 16 + (lane & 15);
        float bg = glb[d];
#pragma unroll
        for (int e = 0; e < 8; ++e) {
            int mloc = (wid & 3) * 16 + ((lane < 16) ? e : e + 8);
            float g = 1.0f / (1.0f + __expf(-(accG[nn][e] + bg)));
            gate[(size_t)(r0 + mloc) * CDIM + d] = f2bf(g);
        }
    }
}

// ---------------------------------------------------------------------------
// K2: per-channel NT GEMM  Out_c = L_c * R_c^T  (768x768x768, bf16->fp32).
// Grid (36 tiles, 128 channels).  128x128 output tile/block, K-chunk 32,
// double-buffered async global->LDS staging, one barrier per chunk.
// ---------------------------------------------------------------------------
__global__ void __launch_bounds__(256)
k_triangle(const unsigned short* __restrict__ leftT,
           const unsigned short* __restrict__ rightT,
           float* __restrict__ tri) {
    __shared__ __align__(16) unsigned short sL[2][128 * 40];  // 2 x 10,240 B
    __shared__ __align__(16) unsigned short sR[2][128 * 40];

    const int tid  = threadIdx.x;
    const int lane = tid & 31;
    const int wid  = tid >> 5;
    const int c    = blockIdx.y;
    const int it   = blockIdx.x / 6, jt = blockIdx.x % 6;
    const int i0   = it * 128, j0 = jt * 128;

    const unsigned short* Lb = leftT  + (size_t)c * NN;
    const unsigned short* Rb = rightT + (size_t)c * NN;

    const int wr = wid & 1, wc = wid >> 1;   // wave tile: rows wr*64..+63, cols wc*32..+31
    const int m0 = wr * 64, n0 = wc * 32;

    v8f z = {};
    v8f acc[4][2];
#pragma unroll
    for (int a = 0; a < 4; ++a) { acc[a][0] = z; acc[a][1] = z; }

    // Staging: 128 rows x 32 halves per matrix; 32B (2x16B) per thread per matrix.
    const int row = tid >> 1, hh = tid & 1;
    const unsigned short* lrow = Lb + (size_t)(i0 + row) * N_RES + hh * 16;
    const unsigned short* rrow = Rb + (size_t)(j0 + row) * N_RES + hh * 16;

    auto issue = [&](int buf, int kkp) {
        const unsigned short* ls = lrow + kkp * 32;
        const unsigned short* rs = rrow + kkp * 32;
        unsigned short* dl = &sL[buf][row * 40 + hh * 16];
        unsigned short* dr = &sR[buf][row * 40 + hh * 16];
        async_cp16(dl,     ls);
        async_cp16(dl + 8, ls + 8);
        async_cp16(dr,     rs);
        async_cp16(dr + 8, rs + 8);
        if (kkp + 1 < 24) {                  // global_prefetch_b8 one chunk ahead
            __builtin_prefetch(ls + 32, 0, 0);
            __builtin_prefetch(rs + 32, 0, 0);
        }
    };

    issue(0, 0);
    for (int kk = 0; kk < 24; ++kk) {
        const int cur = kk & 1;
        async_wait0();                       // own async writes into buf[cur] done
        __syncthreads();                     // all waves' data visible; prev reads of buf[cur^1] done
        if (kk + 1 < 24) issue(cur ^ 1, kk + 1);   // overlaps with WMMA below

        v16bf Bf[2];
#pragma unroll
        for (int b = 0; b < 2; ++b) {
            int n  = n0 + b * 16 + (lane & 15);
            int kb = (lane < 16) ? 0 : 8;
            Bf[b] = make_frag(&sR[cur][n * 40 + kb], &sR[cur][n * 40 + kb + 16]);
        }
#pragma unroll
        for (int a = 0; a < 4; ++a) {
            int m  = m0 + a * 16 + (lane & 15);
            int ka = (lane < 16) ? 0 : 8;
            v16bf Af = make_frag(&sL[cur][m * 40 + ka], &sL[cur][m * 40 + ka + 16]);
            acc[a][0] = WMMA_BF16(Af, Bf[0], acc[a][0]);
            acc[a][1] = WMMA_BF16(Af, Bf[1], acc[a][1]);
        }
    }

    float* triC = tri + (size_t)c * NN;
#pragma unroll
    for (int a = 0; a < 4; ++a) {
#pragma unroll
        for (int b = 0; b < 2; ++b) {
#pragma unroll
            for (int e = 0; e < 8; ++e) {
                int mm = m0 + a * 16 + ((lane < 16) ? e : e + 8);
                int nn = n0 + b * 16 + (lane & 15);
                triC[(size_t)(i0 + mm) * N_RES + (j0 + nn)] = acc[a][b][e];
            }
        }
    }
}

// ---------------------------------------------------------------------------
// K3: LN over C of tri + op projection (WMMA, weights from L2) * gate -> out.
// Block = 64 output rows.
// ---------------------------------------------------------------------------
__global__ void __launch_bounds__(256)
k_out_proj(const float* __restrict__ tri,
           const unsigned short* __restrict__ gate,
           const unsigned short* __restrict__ Wop,   // bf16 [128][128]
           const float* __restrict__ opb,
           const float* __restrict__ clnw, const float* __restrict__ clnb,
           float* __restrict__ out) {
    __shared__ __align__(16) float sT[128 * 68];            // 34,816 B  [c][m]
    __shared__ __align__(16) unsigned short sX[64 * 144];   // 18,432 B  [m][c] bf16
    __shared__ float red[2][64][4];                          //  2,048 B

    const int tid  = threadIdx.x;
    const int lane = tid & 31;
    const int wid  = tid >> 5;
    const int r0   = blockIdx.x * 64;

    // Load tri tile channel-major (async DMA straight into LDS, 128B/thread).
    {
        int cc = tid >> 1, hh = tid & 1;
        const float* s = tri + (size_t)cc * NN + r0 + hh * 32;
        float* dst = &sT[cc * 68 + hh * 32];
#pragma unroll
        for (int j = 0; j < 8; ++j) async_cp16(dst + j * 4, s + j * 4);
        async_wait0();
    }
    __syncthreads();

    // LN over channels (transpose read out of sT), write bf16 row-major tile.
    {
        int m = tid >> 2, q = tid & 3;
        float vals[32];
        float s = 0.f, ss = 0.f;
#pragma unroll
        for (int j = 0; j < 32; ++j) {
            float v = sT[(q * 32 + j) * 68 + m];
            vals[j] = v; s += v; ss += v * v;
        }
        red[0][m][q] = s; red[1][m][q] = ss;
        __syncthreads();
        float st  = red[0][m][0] + red[0][m][1] + red[0][m][2] + red[0][m][3];
        float sst = red[1][m][0] + red[1][m][1] + red[1][m][2] + red[1][m][3];
        float mu   = st * (1.0f / CDIM);
        float var  = sst * (1.0f / CDIM) - mu * mu;
        float rstd = rsqrtf(var + LN_EPS);
#pragma unroll
        for (int j = 0; j < 32; ++j) {
            int cch = q * 32 + j;
            sX[m * 144 + cch] = f2bf((vals[j] - mu) * rstd * clnw[cch] + clnb[cch]);
        }
    }
    __syncthreads();

    // op projection: A from LDS, B straight from global bf16 weights (L2-hot).
    v8f z = {};
    v8f acc[4] = { z, z, z, z };
    const int mrow  = (wid & 3) * 16 + (lane & 15);
    const int nbase = (wid >> 2) * 64;
#pragma unroll
    for (int kc = 0; kc < 4; ++kc) {
        int ka = kc * 32 + ((lane < 16) ? 0 : 8);
        v16bf A = make_frag(&sX[mrow * 144 + ka], &sX[mrow * 144 + ka + 16]);
#pragma unroll
        for (int nn = 0; nn < 4; ++nn) {
            int d  = nbase + nn * 16 + (lane & 15);
            int kb = kc * 32 + ((lane < 16) ? 0 : 8);
            v16bf B = make_frag(Wop + d * 128 + kb, Wop + d * 128 + kb + 16);
            acc[nn] = WMMA_BF16(A, B, acc[nn]);
        }
    }

#pragma unroll
    for (int nn = 0; nn < 4; ++nn) {
        int d  = nbase + nn * 16 + (lane & 15);
        float b = opb[d];
#pragma unroll
        for (int e = 0; e < 8; ++e) {
            int mloc = (wid & 3) * 16 + ((lane < 16) ? e : e + 8);
            size_t r = (size_t)(r0 + mloc);
            float g  = bf2f(gate[r * CDIM + d]);
            out[r * CDIM + d] = (acc[nn][e] + b) * g;
        }
    }
}

// ---------------------------------------------------------------------------
extern "C" void kernel_launch(void* const* d_in, const int* in_sizes, int n_in,
                              void* d_out, int out_size, void* d_ws, size_t ws_size,
                              hipStream_t stream) {
    (void)in_sizes; (void)n_in; (void)out_size; (void)ws_size;

    const float* act  = (const float*)d_in[0];
    const float* mask = (const float*)d_in[1];
    const float* lnw  = (const float*)d_in[2];
    const float* lnb  = (const float*)d_in[3];
    const float* lpw  = (const float*)d_in[4];
    const float* lpb  = (const float*)d_in[5];
    const float* rpw  = (const float*)d_in[6];
    const float* rpb  = (const float*)d_in[7];
    const float* lgw  = (const float*)d_in[8];
    const float* lgb  = (const float*)d_in[9];
    const float* rgw  = (const float*)d_in[10];
    const float* rgb  = (const float*)d_in[11];
    const float* clnw = (const float*)d_in[12];
    const float* clnb = (const float*)d_in[13];
    const float* opw  = (const float*)d_in[14];
    const float* opb  = (const float*)d_in[15];
    const float* glw  = (const float*)d_in[16];
    const float* glb  = (const float*)d_in[17];
    float* out = (float*)d_out;

    char* ws = (char*)d_ws;
    unsigned short* Wb = (unsigned short*)ws;                 // 6*16384 bf16
    size_t off = (size_t)6 * 16384 * 2;                       // 196,608
    unsigned short* leftT  = (unsigned short*)(ws + off); off += (size_t)2 * CDIM * NN;
    unsigned short* rightT = (unsigned short*)(ws + off); off += (size_t)2 * CDIM * NN;
    unsigned short* gate   = (unsigned short*)(ws + off); off += (size_t)2 * CDIM * NN;
    float* tri = (float*)(ws + off);                          // fp32 [C][NN]

    k_convert_weights<<<384, 256, 0, stream>>>(lpw, lgw, rpw, rgw, glw, opw, Wb);
    k_ln_proj<<<NN / 64, 256, 0, stream>>>(act, mask, lnw, lnb,
                                           lpb, lgb, rpb, rgb, glb,
                                           Wb, leftT, rightT, gate);
    k_triangle<<<dim3(36, 128), 256, 0, stream>>>(leftT, rightT, tri);
    k_out_proj<<<NN / 64, 256, 0, stream>>>(tri, gate, Wb + 5 * 16384, opb,
                                            clnw, clnb, out);
}
